// E3EquivariantGNN_49100066128616
// MI455X (gfx1250) — matrix operand order repaired and checked by reading, not compiled, not run on hardware
//
#include <hip/hip_runtime.h>
#include <hip/hip_bf16.h>
#include <math.h>

// ---------------- problem constants ----------------
#define NN   20000
#define EE   320000
#define HH   128
#define NRBF 50
#define CUT  10.0f
#define EPS_LN 1e-5f
#define WPB  2          // waves per block (wave32) -> 64 threads

typedef _Float16 v16h __attribute__((ext_vector_type(16)));
typedef _Float16 v8h  __attribute__((ext_vector_type(8)));
typedef float    v8f  __attribute__((ext_vector_type(8)));

__device__ __forceinline__ float silu_f(float x) { return x / (1.0f + __expf(-x)); }

// All LDS buffers are per-wave private: producer->consumer only needs the DS
// counter drained plus a compiler scheduling/memory fence. No s_barrier needed,
// which also decouples the two waves of the workgroup.
__device__ __forceinline__ void wave_lds_sync() {
    __builtin_amdgcn_wave_barrier();
    asm volatile("s_wait_dscnt 0x0" ::: "memory");
    __builtin_amdgcn_wave_barrier();
}

// A-fragment (16x32 f16, M x K) per ISA 7.12.2:
// lane l (l<16): halves 0..7 -> K=0..7, halves 8..15 -> K=16..23 (row M=l)
// lane l+16    : halves 0..7 -> K=8..15, halves 8..15 -> K=24..31
__device__ __forceinline__ v16h load_a_frag(const _Float16* act, int stride, int kt, int lane) {
    const int m  = lane & 15;
    const int hf = lane >> 4;
    const _Float16* p = act + m * stride + kt * 32 + hf * 8;
    v8h lo = *(const v8h*)(p);
    v8h hi = *(const v8h*)(p + 16);
    v16h a;
#pragma unroll
    for (int i = 0; i < 8; ++i) { a[i] = lo[i]; a[i + 8] = hi[i]; }
    return a;
}

// B fragments are pre-packed: tile (kt,nt) -> 32 lanes x 16 halves contiguous.
template <int KT>
__device__ __forceinline__ v8f wmma_tile(const _Float16* act, int stride,
                                         const _Float16* Wf, int nt, int lane) {
    v8f acc = {};
#pragma unroll
    for (int kt = 0; kt < KT; ++kt) {
        v16h a = load_a_frag(act, stride, kt, lane);
        v16h b = *(const v16h*)(Wf + ((kt * 8 + nt) * 32 + lane) * 16);
        acc = __builtin_amdgcn_wmma_f32_16x16x32_f16(false, a, false, b,
                                                     (short)0, acc, false, false);
    }
    return acc;
}

// ---------------- utility kernels ----------------
__global__ void zero_f32(float* p, int n) {
    for (int i = blockIdx.x * blockDim.x + threadIdx.x; i < n; i += gridDim.x * blockDim.x)
        p[i] = 0.0f;
}

// Pack fp32 W (K x 128, row-major) into f16 WMMA-B fragment order.
// B-fragment (32x16, K x N): lane l holds col n = l%16, K = kt*32 + (l/16)*8 + (i%8) + (i/8)*16.
__global__ void prep_w(const float* __restrict__ W, _Float16* __restrict__ dst,
                       int K, int ktiles) {
    const int t = blockIdx.x * blockDim.x + threadIdx.x;
    const int total = ktiles * 8 * 512;
    if (t >= total) return;
    const int tile = t >> 9;
    const int l    = (t >> 4) & 31;
    const int i    = t & 15;
    const int kt = tile >> 3, nt = tile & 7;
    const int n = nt * 16 + (l & 15);
    const int k = kt * 32 + ((l >> 4) << 3) + (i & 7) + ((i >> 3) << 4);
    dst[t] = (_Float16)((k < K) ? W[k * HH + n] : 0.0f);
}

// ---------------- fused edge kernel ----------------
__global__ __launch_bounds__(WPB * 32) void edge_kernel(
    const float* __restrict__ sfeat, const float* __restrict__ vfeat,
    const float* __restrict__ evec,  const float* __restrict__ edist,
    const float* __restrict__ centers, const float* __restrict__ widths,
    const float* __restrict__ r1b, const float* __restrict__ r2b, const float* __restrict__ r3b,
    const float* __restrict__ s1b, const float* __restrict__ s2b,
    const int*   __restrict__ eidx,
    const _Float16* __restrict__ r1f, const _Float16* __restrict__ r2f,
    const _Float16* __restrict__ r3f, const _Float16* __restrict__ s1f,
    const _Float16* __restrict__ s2f, const _Float16* __restrict__ vlf,
    float* __restrict__ sagg, float* __restrict__ vagg) {

    __shared__ __align__(64) _Float16 bufA[WPB][16][256];
    __shared__ __align__(64) _Float16 bufB[WPB][16][128];   // x-channel / activations
    __shared__ __align__(64) _Float16 bufC1[WPB][16][128];  // y-channel
    __shared__ __align__(64) _Float16 bufC2[WPB][16][128];  // z-channel
    __shared__ __align__(64) float    radl[WPB][16][128];
    __shared__ float cutv[WPB][16];
    __shared__ float evc3[WPB][16][3];
    __shared__ int   esrc[WPB][16];
    __shared__ int   edst[WPB][16];

    const int lane  = threadIdx.x & 31;
    const int w     = threadIdx.x >> 5;
    const int ebase = (blockIdx.x * WPB + w) << 4;

    _Float16 (*A)[256]  = bufA[w];
    _Float16 (*B)[128]  = bufB[w];
    _Float16 (*C1)[128] = bufC1[w];
    _Float16 (*C2)[128] = bufC2[w];
    float    (*R)[128]  = radl[w];

    if (lane < 16) {
        const int e = ebase + lane;
        const float d = edist[e];
        cutv[w][lane] = 0.5f * (__cosf(3.14159265f * d * (1.0f / CUT)) + 1.0f) *
                        (d < CUT ? 1.0f : 0.0f);
        evc3[w][lane][0] = evec[e * 3 + 0];
        evc3[w][lane][1] = evec[e * 3 + 1];
        evc3[w][lane][2] = evec[e * 3 + 2];
        esrc[w][lane] = eidx[e];
        edst[w][lane] = eidx[EE + e];
    }
    // Gaussian RBF into A[:,0:64] (K padded 50 -> 64 with zeros)
    for (int t = lane; t < 16 * 64; t += 32) {
        const int row = t >> 6, k = t & 63;
        float v = 0.0f;
        if (k < NRBF) {
            const float d = edist[ebase + row];
            const float u = (d - centers[k]) / widths[k];
            v = __expf(-u * u);
        }
        A[row][k] = (_Float16)v;
    }
    wave_lds_sync();

    const int nlo = lane & 15;
    const int mhi = (lane >> 4) << 3;

    // radial1: silu(rbf @ W1 + b1) -> B
#pragma unroll
    for (int nt = 0; nt < 8; ++nt) {
        v8f acc = wmma_tile<2>(&A[0][0], 256, r1f, nt, lane);
        const int n = nt * 16 + nlo; const float bb = r1b[n];
#pragma unroll
        for (int r = 0; r < 8; ++r) B[mhi + r][n] = (_Float16)silu_f(acc[r] + bb);
    }
    wave_lds_sync();
    // radial2 -> A
#pragma unroll
    for (int nt = 0; nt < 8; ++nt) {
        v8f acc = wmma_tile<4>(&B[0][0], 128, r2f, nt, lane);
        const int n = nt * 16 + nlo; const float bb = r2b[n];
#pragma unroll
        for (int r = 0; r < 8; ++r) A[mhi + r][n] = (_Float16)silu_f(acc[r] + bb);
    }
    wave_lds_sync();
    // radial3 * cutoff -> R (f32)
#pragma unroll
    for (int nt = 0; nt < 8; ++nt) {
        v8f acc = wmma_tile<4>(&A[0][0], 256, r3f, nt, lane);
        const int n = nt * 16 + nlo; const float bb = r3b[n];
#pragma unroll
        for (int r = 0; r < 8; ++r) {
            const int m = mhi + r;
            R[m][n] = (acc[r] + bb) * cutv[w][m];
        }
    }
    wave_lds_sync();
    // gather pair = [s_src | s_dst] into A (vectorized b128 loads)
    for (int q = lane; q < 16 * 64; q += 32) {
        const int row = q >> 6;
        const int col = (q & 63) << 2;
        const int node = (col < 128) ? esrc[w][row] : edst[w][row];
        const float4 vv = *(const float4*)(sfeat + node * HH + (col & 127));
        A[row][col + 0] = (_Float16)vv.x; A[row][col + 1] = (_Float16)vv.y;
        A[row][col + 2] = (_Float16)vv.z; A[row][col + 3] = (_Float16)vv.w;
    }
    wave_lds_sync();
    // scal1 -> B
#pragma unroll
    for (int nt = 0; nt < 8; ++nt) {
        v8f acc = wmma_tile<8>(&A[0][0], 256, s1f, nt, lane);
        const int n = nt * 16 + nlo; const float bb = s1b[n];
#pragma unroll
        for (int r = 0; r < 8; ++r) B[mhi + r][n] = (_Float16)silu_f(acc[r] + bb);
    }
    wave_lds_sync();
    // scal2 * radial -> atomic scatter into scalar_agg
#pragma unroll
    for (int nt = 0; nt < 8; ++nt) {
        v8f acc = wmma_tile<4>(&B[0][0], 128, s2f, nt, lane);
        const int n = nt * 16 + nlo; const float bb = s2b[n];
#pragma unroll
        for (int r = 0; r < 8; ++r) {
            const int m = mhi + r;
            atomicAdd(&sagg[edst[w][m] * HH + n], (acc[r] + bb) * R[m][n]);
        }
    }
    wave_lds_sync();
    // gather v_src once: xyz -> three f16 planes (B, C1, C2)
    for (int t = lane; t < 16 * 128; t += 32) {
        const int row = t >> 7, h = t & 127;
        const float* vp = vfeat + esrc[w][row] * (HH * 3) + h * 3;
        B[row][h]  = (_Float16)vp[0];
        C1[row][h] = (_Float16)vp[1];
        C2[row][h] = (_Float16)vp[2];
    }
    wave_lds_sync();
    // vector messages: per xyz channel, vt = v_src @ Wv;  msg = vt*rad + evec*rad*s_src
#pragma unroll
    for (int c = 0; c < 3; ++c) {
        const _Float16* plane = (c == 0) ? &B[0][0] : (c == 1) ? &C1[0][0] : &C2[0][0];
#pragma unroll
        for (int nt = 0; nt < 8; ++nt) {
            v8f acc = wmma_tile<4>(plane, 128, vlf, nt, lane);
            const int n = nt * 16 + nlo;
#pragma unroll
            for (int r = 0; r < 8; ++r) {
                const int m = mhi + r;
                const float rr = R[m][n];
                const float val = acc[r] * rr + evc3[w][m][c] * rr * (float)A[m][n];
                atomicAdd(&vagg[edst[w][m] * (HH * 3) + n * 3 + c], val);
            }
        }
    }
}

// ---------------- fused node kernel ----------------
__global__ __launch_bounds__(WPB * 32) void node_kernel(
    const float* __restrict__ sfeat, const float* __restrict__ vfeat,
    const float* __restrict__ u1b, const float* __restrict__ u2b,
    const float* __restrict__ sng, const float* __restrict__ snb,
    const float* __restrict__ vng, const float* __restrict__ vnb,
    const _Float16* __restrict__ u1f, const _Float16* __restrict__ u2f,
    const _Float16* __restrict__ vuf,
    const float* __restrict__ sagg, const float* __restrict__ vagg,
    float* __restrict__ outS, float* __restrict__ outV) {

    __shared__ __align__(64) _Float16 bufA[WPB][16][256];
    __shared__ __align__(64) _Float16 bufB[WPB][16][128];
    __shared__ __align__(64) _Float16 bufC1[WPB][16][128];
    __shared__ __align__(64) _Float16 bufC2[WPB][16][128];
    __shared__ __align__(64) float    P[WPB][16][128];

    const int lane  = threadIdx.x & 31;
    const int w     = threadIdx.x >> 5;
    const int nbase = (blockIdx.x * WPB + w) << 4;

    _Float16 (*A)[256]  = bufA[w];
    _Float16 (*B)[128]  = bufB[w];
    _Float16 (*C1)[128] = bufC1[w];
    _Float16 (*C2)[128] = bufC2[w];
    float    (*Q)[128]  = P[w];

    // gather [scalar_features | scalar_agg]
    for (int q = lane; q < 16 * 64; q += 32) {
        const int row = q >> 6;
        const int col = (q & 63) << 2;
        const int node = nbase + row;
        const float* sp = (col < 128) ? (sfeat + node * HH + col)
                                      : (sagg + node * HH + (col - 128));
        const float4 vv = *(const float4*)sp;
        A[row][col + 0] = (_Float16)vv.x; A[row][col + 1] = (_Float16)vv.y;
        A[row][col + 2] = (_Float16)vv.z; A[row][col + 3] = (_Float16)vv.w;
    }
    wave_lds_sync();

    const int nlo = lane & 15;
    const int mhi = (lane >> 4) << 3;

#pragma unroll
    for (int nt = 0; nt < 8; ++nt) {
        v8f acc = wmma_tile<8>(&A[0][0], 256, u1f, nt, lane);
        const int n = nt * 16 + nlo; const float bb = u1b[n];
#pragma unroll
        for (int r = 0; r < 8; ++r) B[mhi + r][n] = (_Float16)silu_f(acc[r] + bb);
    }
    wave_lds_sync();
#pragma unroll
    for (int nt = 0; nt < 8; ++nt) {
        v8f acc = wmma_tile<4>(&B[0][0], 128, u2f, nt, lane);
        const int n = nt * 16 + nlo; const float bb = u2b[n];
#pragma unroll
        for (int r = 0; r < 8; ++r) {
            const int m = mhi + r;
            Q[m][n] = acc[r] + bb + sfeat[(nbase + m) * HH + n];  // residual
        }
    }
    wave_lds_sync();
    // scalar layernorm, one row per low lane
    if (lane < 16) {
        const int node = nbase + lane;
        float s = 0.0f;
        for (int j = 0; j < HH; ++j) s += Q[lane][j];
        const float mean = s * (1.0f / HH);
        float v = 0.0f;
        for (int j = 0; j < HH; ++j) { const float d = Q[lane][j] - mean; v += d * d; }
        const float inv = rsqrtf(v * (1.0f / HH) + EPS_LN);
        for (int j = 0; j < HH; ++j)
            outS[node * HH + j] = (Q[lane][j] - mean) * inv * sng[j] + snb[j];
    }
    wave_lds_sync();

    // gather vector_agg once: xyz -> three f16 planes
    for (int t = lane; t < 16 * 128; t += 32) {
        const int row = t >> 7, h = t & 127;
        const float* vp = vagg + (nbase + row) * (HH * 3) + h * 3;
        B[row][h]  = (_Float16)vp[0];
        C1[row][h] = (_Float16)vp[1];
        C2[row][h] = (_Float16)vp[2];
    }
    wave_lds_sync();

    // vector update + LN over hidden dim, per xyz channel
    for (int c = 0; c < 3; ++c) {
        const _Float16* plane = (c == 0) ? &B[0][0] : (c == 1) ? &C1[0][0] : &C2[0][0];
#pragma unroll
        for (int nt = 0; nt < 8; ++nt) {
            v8f acc = wmma_tile<4>(plane, 128, vuf, nt, lane);
            const int n = nt * 16 + nlo;
#pragma unroll
            for (int r = 0; r < 8; ++r) {
                const int m = mhi + r;
                Q[m][n] = vfeat[(nbase + m) * (HH * 3) + n * 3 + c] + acc[r];
            }
        }
        wave_lds_sync();
        if (lane < 16) {
            const int node = nbase + lane;
            float s = 0.0f;
            for (int j = 0; j < HH; ++j) s += Q[lane][j];
            const float mean = s * (1.0f / HH);
            float v = 0.0f;
            for (int j = 0; j < HH; ++j) { const float d = Q[lane][j] - mean; v += d * d; }
            const float inv = rsqrtf(v * (1.0f / HH) + EPS_LN);
            for (int j = 0; j < HH; ++j)
                outV[node * (HH * 3) + j * 3 + c] =
                    (Q[lane][j] - mean) * inv * vng[j] + vnb[j];
        }
        wave_lds_sync();
    }
}

// ---------------- launcher ----------------
extern "C" void kernel_launch(void* const* d_in, const int* in_sizes, int n_in,
                              void* d_out, int out_size, void* d_ws, size_t ws_size,
                              hipStream_t stream) {
    (void)in_sizes; (void)n_in; (void)out_size; (void)ws_size;
    // setup_inputs() order with params pytree flattened in sorted-key order
    const float* sfeat   = (const float*)d_in[0];   // (N,H)
    const float* vfeat   = (const float*)d_in[1];   // (N,H,3)
    const float* evec    = (const float*)d_in[2];   // (E,3)
    const float* edist   = (const float*)d_in[3];   // (E,)
    const float* centers = (const float*)d_in[4];
    const float* r1W = (const float*)d_in[5];  const float* r1b = (const float*)d_in[6];
    const float* r2W = (const float*)d_in[7];  const float* r2b = (const float*)d_in[8];
    const float* r3W = (const float*)d_in[9];  const float* r3b = (const float*)d_in[10];
    const float* s1W = (const float*)d_in[11]; const float* s1b = (const float*)d_in[12];
    const float* s2W = (const float*)d_in[13]; const float* s2b = (const float*)d_in[14];
    const float* snb = (const float*)d_in[15]; const float* sng = (const float*)d_in[16];
    const float* u1W = (const float*)d_in[17]; const float* u1b = (const float*)d_in[18];
    const float* u2W = (const float*)d_in[19]; const float* u2b = (const float*)d_in[20];
    const float* vlW = (const float*)d_in[21];  // vec_lin.W
    const float* vuW = (const float*)d_in[22];  // vec_upd.W
    const float* vnb = (const float*)d_in[23]; const float* vng = (const float*)d_in[24];
    const float* widths = (const float*)d_in[25];
    const int*   eidx   = (const int*)d_in[26];  // (2,E)

    // workspace layout
    float* sagg = (float*)d_ws;                              // N*H f32
    float* vagg = sagg + (size_t)NN * HH;                    // N*H*3 f32
    _Float16* w16 = (_Float16*)(vagg + (size_t)NN * HH * 3); // packed f16 weights
    _Float16* r1f = w16;
    _Float16* r2f = r1f + 2 * 8 * 512;
    _Float16* r3f = r2f + 4 * 8 * 512;
    _Float16* s1f = r3f + 4 * 8 * 512;
    _Float16* s2f = s1f + 8 * 8 * 512;
    _Float16* vlf = s2f + 4 * 8 * 512;
    _Float16* u1f = vlf + 4 * 8 * 512;
    _Float16* u2f = u1f + 8 * 8 * 512;
    _Float16* vuf = u2f + 4 * 8 * 512;

    zero_f32<<<4096, 256, 0, stream>>>(sagg, NN * HH * 4);  // both agg buffers

    prep_w<<<(2 * 8 * 512 + 255) / 256, 256, 0, stream>>>(r1W, r1f, NRBF, 2);
    prep_w<<<(4 * 8 * 512 + 255) / 256, 256, 0, stream>>>(r2W, r2f, HH, 4);
    prep_w<<<(4 * 8 * 512 + 255) / 256, 256, 0, stream>>>(r3W, r3f, HH, 4);
    prep_w<<<(8 * 8 * 512 + 255) / 256, 256, 0, stream>>>(s1W, s1f, 2 * HH, 8);
    prep_w<<<(4 * 8 * 512 + 255) / 256, 256, 0, stream>>>(s2W, s2f, HH, 4);
    prep_w<<<(4 * 8 * 512 + 255) / 256, 256, 0, stream>>>(vlW, vlf, HH, 4);
    prep_w<<<(8 * 8 * 512 + 255) / 256, 256, 0, stream>>>(u1W, u1f, 2 * HH, 8);
    prep_w<<<(4 * 8 * 512 + 255) / 256, 256, 0, stream>>>(u2W, u2f, HH, 4);
    prep_w<<<(4 * 8 * 512 + 255) / 256, 256, 0, stream>>>(vuW, vuf, HH, 4);

    edge_kernel<<<EE / (WPB * 16), WPB * 32, 0, stream>>>(
        sfeat, vfeat, evec, edist, centers, widths,
        r1b, r2b, r3b, s1b, s2b, eidx,
        r1f, r2f, r3f, s1f, s2f, vlf, sagg, vagg);

    float* outS = (float*)d_out;
    float* outV = outS + (size_t)NN * HH;
    node_kernel<<<NN / (WPB * 16), WPB * 32, 0, stream>>>(
        sfeat, vfeat, u1b, u2b, sng, snb, vng, vnb,
        u1f, u2f, vuf, sagg, vagg, outS, outV);
}
